// NeuralNetwork_84069689851935
// MI455X (gfx1250) — matrix operand order repaired and checked by reading, not compiled, not run on hardware
//
#include <hip/hip_runtime.h>
#include <hip/hip_bf16.h>
#include <math.h>

// ---------------------------------------------------------------------------
// Persistent-RNN for MI455X (gfx1250, wave32, WMMA, async global->LDS).
//
//  - 192 workgroups x 128 threads (4 wave32). WG g owns output rows
//    [16g, 16g+16) of W1, W2, W3, kept resident in LDS as bf16 in the
//    WMMA A-fragment swizzled layout (3 x 96 KB = 288 KB of 320 KB).
//  - Inter-layer activations live in the workspace as bf16 and are staged
//    into LDS with GLOBAL_LOAD_ASYNC_TO_LDS_B128 (ASYNCcnt), so the per-step
//    critical path is: async 6 KB L2->LDS copy -> 96 v_wmma_f32_16x16x32_bf16
//    (K split over 4 waves) -> LDS reduce -> bias/tanh -> 32 B publish ->
//    device-wide barrier.  3 barriers / step, 4096 steps.
// ---------------------------------------------------------------------------

typedef __attribute__((ext_vector_type(16))) __bf16 v16bf;
typedef __attribute__((ext_vector_type(8)))  float  v8f;

#define VOCAB    1024
#define STATE_N  2048
#define CDIM     3072
#define TSTEPS   4096
#define NBLK     192          // CDIM / 16 row-groups
#define NTHR     128          // 4 wave32
#define NWAVE    4
#define KCHUNKS  96           // CDIM / 32
#define CPW      (KCHUNKS / NWAVE)

#define W_SLICE_HALFS (KCHUNKS * 32 * 16)        // 49152 bf16 = 96 KB / matrix
#define LDS_W_BYTES   (3 * W_SLICE_HALFS * 2)    // 288 KB
#define LDS_H_OFF     LDS_W_BYTES
#define LDS_H_BYTES   (CDIM * 2)                 // 6 KB activation (bf16)
#define LDS_RED_OFF   (LDS_H_OFF + LDS_H_BYTES)
#define LDS_RED_BYTES (NTHR * 8 * 4)             // 4 KB reduction scratch
#define LDS_TOTAL     (LDS_RED_OFF + LDS_RED_BYTES)   // 305152 B <= 320 KB

// ---- workspace layout (bytes) ---------------------------------------------
// [0,      6144)   activation buffer A   (3072 x bf16)
// [6144,  12288)   activation buffer B   (3072 x bf16)
// [12288, 16384)   recurrent state       (2048 x bf16)
// [16384, 16392)   grid barrier {count, generation} (unsigned[2])
#define WSB_ACTA  0
#define WSB_ACTB  6144
#define WSB_STATE 12288
#define WSB_BAR   16384

// -------- gfx1250 async global->LDS copy (ASYNCcnt) ------------------------
__device__ __forceinline__ void async_g2l_b128(void* lds_dst, const void* gsrc) {
    // Low 32 bits of a generic LDS pointer are the wave-relative LDS address
    // (flat->LDS mapping truncates to addr[31:0]).
    unsigned l = (unsigned)(uintptr_t)lds_dst;
    asm volatile("global_load_async_to_lds_b128 %0, %1, off"
                 :: "v"(l), "v"(gsrc) : "memory");
}
__device__ __forceinline__ void wait_async0() {
    asm volatile("s_wait_asynccnt 0" ::: "memory");
}

// K-offset within a 32-chunk for half t of the 16-bit WMMA A fragment
// (contiguous order k = 16*(lane/16) + t, matching the B fill).
__device__ __forceinline__ int a_koff(int lane, int t) {
    return (lane < 16) ? t : t + 16;
}

// Convert 16 rows [r0, r0+16) of a CDIM x CDIM f32 matrix into the bf16
// WMMA A-fragment-swizzled LDS image: slot (chunk c, lane l) holds the 16
// contiguous halves lane l consumes for that K-chunk.
__device__ __forceinline__ void load_weight_slice(const float* __restrict__ Wg,
                                                  int r0, __bf16* dst) {
    const int tid = threadIdx.x;
    for (int idx = tid; idx < KCHUNKS * 32; idx += NTHR) {
        const int c     = idx >> 5;
        const int l     = idx & 31;
        const int m     = l & 15;               // matrix row within the tile
        const int kbase = c << 5;
        const float* row = Wg + (size_t)(r0 + m) * CDIM;
        __bf16* o = dst + idx * 16;             // 32 B aligned
#pragma unroll
        for (int t = 0; t < 16; ++t)
            o[t] = (__bf16)row[kbase + a_koff(l, t)];
    }
}

// 16-row GEMV: y[m] = sum_k Wslice[m,k] * h[k], K split over 4 waves,
// accumulated with v_wmma_f32_16x16x32_bf16, reduced through LDS.
// Valid result only for tid < 16 (row index = tid).
__device__ __forceinline__ float gemv16(const __bf16* __restrict__ wslice,
                                        const __bf16* __restrict__ h,
                                        float* __restrict__ red) {
    const int tid  = threadIdx.x;
    const int lane = tid & 31;
    const int wave = tid >> 5;

    v8f acc = {};
    const int c0 = wave * CPW;
#pragma unroll 4
    for (int c = c0; c < c0 + CPW; ++c) {
        // A: pre-swizzled weights (16 contiguous halves / lane)
        v16bf a = *(const v16bf*)(wslice + ((c << 5) + lane) * 16);
        // B: activation chunk broadcast into all 16 columns.
        // lanes 0-15 read h[32c .. 32c+15], lanes 16-31 read h[32c+16 ..].
        v16bf b = *(const v16bf*)(h + (c << 5) + (lane & 16));
        acc = __builtin_amdgcn_wmma_f32_16x16x32_bf16(
                  false, a, false, b, (short)0, acc, false, false);
    }

    // D layout: acc[d] on lanes 0-15 carries row d; lanes 16-31 carry row d+8
    // (all columns identical since B was column-broadcast).
#pragma unroll
    for (int d = 0; d < 8; ++d) red[tid * 8 + d] = acc[d];
    __syncthreads();

    float y = 0.f;
    if (tid < 16) {
        const int lsel = (tid < 8) ? 0 : 16;
        const int d    = tid & 7;
#pragma unroll
        for (int w = 0; w < NWAVE; ++w) y += red[(w * 32 + lsel) * 8 + d];
    }
    __syncthreads();   // red & h are reusable after this
    return y;
}

// Sense-reversing device-wide barrier on {count, generation} in global ws.
__device__ __forceinline__ void grid_barrier(unsigned* cnt, unsigned* gen) {
    __threadfence();
    __syncthreads();
    if (threadIdx.x == 0) {
        volatile unsigned* vgen = gen;
        const unsigned g = *vgen;
        const unsigned v = atomicAdd(cnt, 1u);
        if (v == (unsigned)(NBLK - 1)) {
            *cnt = 0u;
            __threadfence();
            atomicAdd(gen, 1u);
        } else {
            while (*vgen == g) __builtin_amdgcn_s_sleep(2);
        }
    }
    __syncthreads();
    __threadfence();
}

__global__ void rnn_init(const float* __restrict__ s0, char* __restrict__ ws) {
    const int i = blockIdx.x * blockDim.x + threadIdx.x;
    __bf16* st = (__bf16*)(ws + WSB_STATE);
    if (i < STATE_N) st[i] = (__bf16)s0[i];
    if (i == 0) {
        unsigned* bar = (unsigned*)(ws + WSB_BAR);
        bar[0] = 0u;
        bar[1] = 0u;
    }
}

__global__ void __launch_bounds__(NTHR)
rnn_persist(const float* __restrict__ x,
            const float* __restrict__ W1, const float* __restrict__ B1,
            const float* __restrict__ W2, const float* __restrict__ B2,
            const float* __restrict__ W3, const float* __restrict__ B3,
            float* __restrict__ out, char* __restrict__ ws) {
    extern __shared__ __align__(32) char smem[];
    __bf16* w1s = (__bf16*)smem;
    __bf16* w2s = w1s + W_SLICE_HALFS;
    __bf16* w3s = w2s + W_SLICE_HALFS;
    __bf16* hld = (__bf16*)(smem + LDS_H_OFF);
    float*  red = (float*)(smem + LDS_RED_OFF);

    char* actA = ws + WSB_ACTA;                 // bf16[3072]
    char* actB = ws + WSB_ACTB;                 // bf16[3072]
    __bf16* st = (__bf16*)(ws + WSB_STATE);     // bf16[2048]
    unsigned* bar = (unsigned*)(ws + WSB_BAR);

    const int g   = blockIdx.x;
    const int r0  = g << 4;
    const int tid = threadIdx.x;

    // One-time weight residency: HBM -> bf16 -> swizzled LDS (288 KB).
    load_weight_slice(W1, r0, w1s);
    load_weight_slice(W2, r0, w2s);
    load_weight_slice(W3, r0, w3s);
    __syncthreads();
    grid_barrier(bar, bar + 1);   // also orders init-kernel state copy

    float* y_out = out;                               // [TSTEPS][VOCAB]
    float* s_out = out + (size_t)TSTEPS * VOCAB;      // final state (2048)

    for (int t = 0; t < TSTEPS; ++t) {
        // ---- layer 1: inp = concat(x_t, state) ----------------------------
        const float* xt = x + (size_t)t * VOCAB;
        // state half: pure bf16 copy, async global->LDS (4 KB, ASYNCcnt)
#pragma unroll
        for (int off = tid * 16; off < STATE_N * 2; off += NTHR * 16)
            async_g2l_b128((char*)(hld + VOCAB) + off, (char*)st + off);
        // x half: f32 input -> bf16 (only per-step conversion left)
        for (int k = tid; k < VOCAB; k += NTHR)
            hld[k] = (__bf16)xt[k];
        // keep next step's x slice marching HBM -> L2 while we compute
        if (t + 1 < TSTEPS)
            __builtin_prefetch((const char*)(x + (size_t)(t + 1) * VOCAB) + tid * 32, 0, 1);
        wait_async0();
        __syncthreads();
        const float y1 = gemv16(w1s, hld, red);
        if (tid < 16)
            *((__bf16*)actA + r0 + tid) = (__bf16)tanhf(y1 + B1[r0 + tid]);
        grid_barrier(bar, bar + 1);

        // ---- layer 2 ------------------------------------------------------
#pragma unroll
        for (int off = tid * 16; off < CDIM * 2; off += NTHR * 16)
            async_g2l_b128((char*)hld + off, actA + off);
        wait_async0();
        __syncthreads();
        const float y2 = gemv16(w2s, hld, red);
        if (tid < 16)
            *((__bf16*)actB + r0 + tid) = (__bf16)tanhf(y2 + B2[r0 + tid]);
        grid_barrier(bar, bar + 1);

        // ---- layer 3 (no tanh): rows <1024 -> logits, >=1024 -> state -----
#pragma unroll
        for (int off = tid * 16; off < CDIM * 2; off += NTHR * 16)
            async_g2l_b128((char*)hld + off, actB + off);
        wait_async0();
        __syncthreads();
        const float y3 = gemv16(w3s, hld, red);
        if (tid < 16) {
            const int r = r0 + tid;
            const float v = y3 + B3[r];
            if (r < VOCAB) {
                y_out[(size_t)t * VOCAB + r] = v;          // logits stay f32
            } else {
                st[r - VOCAB] = (__bf16)v;                 // recurrent path
                if (t == TSTEPS - 1) s_out[r - VOCAB] = v; // final state f32
            }
        }
        grid_barrier(bar, bar + 1);
    }
}

extern "C" void kernel_launch(void* const* d_in, const int* in_sizes, int n_in,
                              void* d_out, int out_size, void* d_ws, size_t ws_size,
                              hipStream_t stream) {
    const float* x  = (const float*)d_in[0];
    const float* s0 = (const float*)d_in[1];
    const float* W1 = (const float*)d_in[2];
    const float* B1 = (const float*)d_in[3];
    const float* W2 = (const float*)d_in[4];
    const float* B2 = (const float*)d_in[5];
    const float* W3 = (const float*)d_in[6];
    const float* B3 = (const float*)d_in[7];
    char* ws = (char*)d_ws;

    rnn_init<<<dim3((STATE_N + 255) / 256), dim3(256), 0, stream>>>(s0, ws);
    rnn_persist<<<dim3(NBLK), dim3(NTHR), LDS_TOTAL, stream>>>(
        x, W1, B1, W2, B2, W3, B3, (float*)d_out, ws);
}
